// SelfAttention_42305427865792
// MI455X (gfx1250) — compile-verified, matrix operand order checked by reading, not run on hardware
//
#include <hip/hip_runtime.h>
#include <stdint.h>
#include <math.h>

// ---------------------------------------------------------------------------
// MI455X (gfx1250) fused self-attention, bf16 WMMA + TDM pipeline:
//   1) gemm_qkv_bf16 : qkv = x @ w_in^T + b_in, epilogue writes bf16
//                      q|k rows (4096 x 2048) and V pre-TRANSPOSED per head
//                      (vT[b,h][d][s]) so attention needs no VALU transpose.
//   2) flash_attn    : online-softmax attention; K/V tiles streamed by the
//                      Tensor Data Mover (tensor_load_to_lds, double-buffered
//                      on TENSORcnt), all matmuls on v_wmma_f32_16x16x32_bf16.
//   3) gemm_out_f32  : out = attn @ w_out^T + b_out (fp32 out).
// ---------------------------------------------------------------------------

typedef __attribute__((ext_vector_type(16))) __bf16 v16bf;
typedef __attribute__((ext_vector_type(8)))  __bf16 v8bf;
typedef __attribute__((ext_vector_type(4)))  __bf16 v4bf;
typedef __attribute__((ext_vector_type(8)))  float  v8f;
typedef __attribute__((ext_vector_type(4)))  float  v4f;
typedef __attribute__((ext_vector_type(4)))  unsigned int u32x4;
typedef __attribute__((ext_vector_type(8)))  int    i32x8;
typedef __attribute__((ext_vector_type(4)))  int    i32x4;

// 16x32 bf16 WMMA A/B fragment from K-contiguous storage (LDS or global).
// ISA 7.12.2: lanes 0-15 hold K quarters {0,2}, lanes 16-31 hold {1,3};
// caller passes p = rowbase + 8*halfsel (+32*kstep), 16B aligned.
__device__ __forceinline__ v16bf frag_ld(const __bf16* p) {
  v8bf lo = *(const v8bf*)(p);
  v8bf hi = *(const v8bf*)(p + 16);
  return __builtin_shufflevector(lo, hi, 0, 1, 2, 3, 4, 5, 6, 7,
                                         8, 9, 10, 11, 12, 13, 14, 15);
}

__device__ __forceinline__ v8f wmma_bf16(v16bf a, v16bf b, v8f c) {
  return __builtin_amdgcn_wmma_f32_16x16x32_bf16(false, a, false, b,
                                                 (short)0, c, false, false);
}

__device__ __forceinline__ float hmax16(float v) {
  v = fmaxf(v, __shfl_xor(v, 1, 32));
  v = fmaxf(v, __shfl_xor(v, 2, 32));
  v = fmaxf(v, __shfl_xor(v, 4, 32));
  v = fmaxf(v, __shfl_xor(v, 8, 32));
  return v;
}
__device__ __forceinline__ float hsum16(float v) {
  v += __shfl_xor(v, 1, 32);
  v += __shfl_xor(v, 2, 32);
  v += __shfl_xor(v, 4, 32);
  v += __shfl_xor(v, 8, 32);
  return v;
}

// ---------------------------------------------------------------------------
// TDM: load a 64x64 bf16 tile (row stride 2048 elements in global) into LDS
// with 16B padding per 128B row (LDS row stride 144B = 72 bf16).
// D# per ISA §8.3/8.4: count=1, type=2, data_size=1(2B), pad_interval=4
// (32 DWORDs=128B), pad_amount=3 (4 DWORDs=16B), tile 64x64, dims 64/64.
// This toolchain's builtin takes 6 args: (g0, g1, g2, g3, extra g, cpol).
// ---------------------------------------------------------------------------
__device__ __forceinline__ void tdm_load_tile64(unsigned lds_addr,
                                                unsigned long long gaddr) {
  u32x4 g0;
  g0[0] = 1u;                                            // count=1, user mode
  g0[1] = lds_addr;                                      // LDS byte address
  g0[2] = (unsigned)gaddr;                               // global_addr[31:0]
  g0[3] = ((unsigned)(gaddr >> 32) & 0x01FFFFFFu) | 0x80000000u;  // type=2
  i32x8 g1;
  g1[0] = (1 << 16) | (1 << 20) | (4 << 22) | (3 << 25); // data_size|pad cfg
  g1[1] = 64 << 16;                                      // tensor_dim0 lo16
  g1[2] = 64 << 16;                                      // tensor_dim1 lo16
  g1[3] = 64 << 16;                                      // tile_dim0
  g1[4] = 64;                                            // tile_dim1
  g1[5] = 2048;                                          // dim0_stride lo32
  g1[6] = 0;
  g1[7] = 0;
  i32x4 z4 = {};
  i32x8 z8 = {};
  __builtin_amdgcn_tensor_load_to_lds(g0, g1, z4, z4, z8, 0);
}

// ---------------------------------------------------------------------------
// Shared WMMA GEMM main loop: acc(64x32 per wave) += A(128x K) @ W^T tiles.
// 256 threads, wave grid 2(M) x 4(N), LDS row stride 40 bf16 = 80B.
// ---------------------------------------------------------------------------
__device__ __forceinline__ void gemm_mainloop(
    const float* __restrict__ A, int lda, const float* __restrict__ W, int ldw,
    int K, int m0, int n0, int wm, int wn, int l16, int hl, int lr, int lc,
    __bf16* ldsA, __bf16* ldsB, v8f acc[4][2]) {
  for (int k0 = 0; k0 < K; k0 += 32) {
    __syncthreads();
#pragma unroll
    for (int p = 0; p < 4; ++p) {
      int row = lr + 32 * p;
      v4f av = *(const v4f*)(A + (size_t)(m0 + row) * lda + k0 + lc);
      v4f wv = *(const v4f*)(W + (size_t)(n0 + row) * ldw + k0 + lc);
      v4bf ab, wb;
#pragma unroll
      for (int i = 0; i < 4; ++i) { ab[i] = (__bf16)av[i]; wb[i] = (__bf16)wv[i]; }
      *(v4bf*)&ldsA[row * 40 + lc] = ab;
      *(v4bf*)&ldsB[row * 40 + lc] = wb;
    }
    __syncthreads();
    v16bf af[4], bf[2];
#pragma unroll
    for (int mt = 0; mt < 4; ++mt)
      af[mt] = frag_ld(&ldsA[(wm * 64 + mt * 16 + l16) * 40 + 8 * hl]);
#pragma unroll
    for (int nt = 0; nt < 2; ++nt)
      bf[nt] = frag_ld(&ldsB[(wn * 32 + nt * 16 + l16) * 40 + 8 * hl]);
#pragma unroll
    for (int mt = 0; mt < 4; ++mt)
#pragma unroll
      for (int nt = 0; nt < 2; ++nt)
        acc[mt][nt] = wmma_bf16(af[mt], bf[nt], acc[mt][nt]);
  }
}

// ---------------------------------------------------------------------------
// Stage 1: qkv projection, bf16 epilogue. n<2048 -> qk rows; n>=2048 -> vT.
// qk: (4096, 2048) bf16.  vt: per (b,h): [64][2048] bf16 (d-major).
// ---------------------------------------------------------------------------
__global__ __launch_bounds__(256) void gemm_qkv_bf16(
    const float* __restrict__ X, const float* __restrict__ Win,
    const float* __restrict__ bin, __bf16* __restrict__ qk,
    __bf16* __restrict__ vt) {
  __shared__ __align__(16) __bf16 ldsA[128 * 40];
  __shared__ __align__(16) __bf16 ldsB[128 * 40];
  const int tid = threadIdx.x, lane = tid & 31, wave = tid >> 5;
  const int hl = lane >> 4, l16 = lane & 15;
  const int wm = wave >> 2, wn = wave & 3;
  const int m0 = blockIdx.y * 128, n0 = blockIdx.x * 128;
  const int lr = tid >> 3, lc = (tid & 7) << 2;

  v8f acc[4][2] = {};
  gemm_mainloop(X, 1024, Win, 1024, 1024, m0, n0, wm, wn, l16, hl, lr, lc,
                ldsA, ldsB, acc);

  if (n0 < 2048) {  // q | k region: row-major bf16, d contiguous
#pragma unroll
    for (int nt = 0; nt < 2; ++nt) {
      int n = n0 + wn * 32 + nt * 16 + l16;
      float bv = bin[n];
#pragma unroll
      for (int mt = 0; mt < 4; ++mt) {
        int mbase = m0 + wm * 64 + mt * 16 + 8 * hl;
#pragma unroll
        for (int r = 0; r < 8; ++r)
          qk[(size_t)(mbase + r) * 2048 + n] = (__bf16)(acc[mt][nt][r] + bv);
      }
    }
  } else {          // v region: write transposed, s contiguous (16B stores)
#pragma unroll
    for (int nt = 0; nt < 2; ++nt) {
      int n = n0 + wn * 32 + nt * 16 + l16;
      float bv = bin[n];
      int nv = n - 2048;
      int h = nv >> 6, d = nv & 63;
#pragma unroll
      for (int mt = 0; mt < 4; ++mt) {
        int mbase = m0 + wm * 64 + mt * 16 + 8 * hl;
        int batch = mbase >> 11, s0 = mbase & 2047;
        v8bf pk;
#pragma unroll
        for (int r = 0; r < 8; ++r) pk[r] = (__bf16)(acc[mt][nt][r] + bv);
        *(v8bf*)&vt[((size_t)(batch * 16 + h) * 64 + d) * 2048 + s0] = pk;
      }
    }
  }
}

// ---------------------------------------------------------------------------
// Stage 3: out = attn @ w_out^T + b_out, fp32 epilogue.
// ---------------------------------------------------------------------------
__global__ __launch_bounds__(256) void gemm_out_f32(
    const float* __restrict__ Ain, const float* __restrict__ Wout,
    const float* __restrict__ bout, float* __restrict__ C) {
  __shared__ __align__(16) __bf16 ldsA[128 * 40];
  __shared__ __align__(16) __bf16 ldsB[128 * 40];
  const int tid = threadIdx.x, lane = tid & 31, wave = tid >> 5;
  const int hl = lane >> 4, l16 = lane & 15;
  const int wm = wave >> 2, wn = wave & 3;
  const int m0 = blockIdx.y * 128, n0 = blockIdx.x * 128;
  const int lr = tid >> 3, lc = (tid & 7) << 2;

  v8f acc[4][2] = {};
  gemm_mainloop(Ain, 1024, Wout, 1024, 1024, m0, n0, wm, wn, l16, hl, lr, lc,
                ldsA, ldsB, acc);
#pragma unroll
  for (int nt = 0; nt < 2; ++nt) {
    int n = n0 + wn * 32 + nt * 16 + l16;
    float bv = bout[n];
#pragma unroll
    for (int mt = 0; mt < 4; ++mt) {
      int mbase = m0 + wm * 64 + mt * 16 + 8 * hl;
#pragma unroll
      for (int r = 0; r < 8; ++r)
        C[(size_t)(mbase + r) * 1024 + n] = acc[mt][nt][r] + bv;
    }
  }
}

// ---------------------------------------------------------------------------
// Stage 2: flash attention. Block = 128 q rows (16/wave) for one (b,h).
// K/V tiles (64 kv rows) DMA'd by TDM into double-buffered padded LDS;
// wave 0 issues next tile then waits TENSORcnt<=2 (tensor ops complete
// in-order per wave => the current tile's two ops are done).
// ---------------------------------------------------------------------------
__global__ __launch_bounds__(256) void flash_attn(
    const __bf16* __restrict__ qk, const __bf16* __restrict__ vt,
    float* __restrict__ out) {
  __shared__ __align__(16) __bf16 ldsK[2][64 * 72];   // [kv][d], stride 144B
  __shared__ __align__(16) __bf16 ldsV[2][64 * 72];   // [d][kv]
  __shared__ __align__(16) __bf16 ldsP[8][16 * 72];   // per-wave P scratch

  const int tid = threadIdx.x, lane = tid & 31, wave = tid >> 5;
  const int hl = lane >> 4, l16 = lane & 15;
  const int bh = blockIdx.y;
  const int batch = bh >> 4, head = bh & 15;
  const size_t srow0 = (size_t)batch * 2048;
  const int q0 = blockIdx.x * 128 + wave * 16;

  // Q fragments straight from bf16 global (no conversion needed).
  const __bf16* qrow = qk + (srow0 + q0 + l16) * 2048 + head * 64;
  v16bf aq[2];
#pragma unroll
  for (int j = 0; j < 2; ++j) aq[j] = frag_ld(qrow + 32 * j + 8 * hl);

  v8f oacc[4] = {};
  float mst[8], lst[8];
#pragma unroll
  for (int r = 0; r < 8; ++r) { mst[r] = -1.0e30f; lst[r] = 0.0f; }

  const unsigned long long gkb =
      (unsigned long long)(uintptr_t)(qk + srow0 * 2048 + 1024 + head * 64);
  const unsigned long long gvb =
      (unsigned long long)(uintptr_t)(vt + (size_t)bh * 64 * 2048);
  const unsigned ldsKa[2] = {(unsigned)(uintptr_t)&ldsK[0][0],
                             (unsigned)(uintptr_t)&ldsK[1][0]};
  const unsigned ldsVa[2] = {(unsigned)(uintptr_t)&ldsV[0][0],
                             (unsigned)(uintptr_t)&ldsV[1][0]};

  if (wave == 0) {  // prologue: tile 0 -> buffer 0
    tdm_load_tile64(ldsKa[0], gkb);
    tdm_load_tile64(ldsVa[0], gvb);
  }

  for (int kv0 = 0; kv0 < 2048; kv0 += 64) {
    const int cur = (kv0 >> 6) & 1;
    __syncthreads();                 // prev compute done; other buffer free
    if (wave == 0) {
      if (kv0 + 64 < 2048) {         // prefetch next tile, K row adv, V col adv
        tdm_load_tile64(ldsKa[cur ^ 1], gkb + (size_t)(kv0 + 64) * 2048 * 2);
        tdm_load_tile64(ldsVa[cur ^ 1], gvb + (size_t)(kv0 + 64) * 2);
        __builtin_amdgcn_s_wait_tensorcnt(2);   // current tile complete
      } else {
        __builtin_amdgcn_s_wait_tensorcnt(0);
      }
    }
    __syncthreads();                 // current tile visible to all waves
    const __bf16* Kt = &ldsK[cur][0];
    const __bf16* Vt = &ldsV[cur][0];

    // S = Q @ K^T (B column n = kv row n -> Kt row n).
    v8f st[4];
#pragma unroll
    for (int nt = 0; nt < 4; ++nt) {
      v8f s = {};
#pragma unroll
      for (int j = 0; j < 2; ++j) {
        v16bf bk = frag_ld(&Kt[(nt * 16 + l16) * 72 + 32 * j + 8 * hl]);
        s = wmma_bf16(aq[j], bk, s);
      }
#pragma unroll
      for (int r = 0; r < 8; ++r) st[nt][r] = s[r] * 0.125f;  // 1/sqrt(64)
    }

    // Online softmax (fp32); row r+8*hl spans the 16 lanes of this half.
    float pr[4][8];
#pragma unroll
    for (int r = 0; r < 8; ++r) {
      float mx = fmaxf(fmaxf(st[0][r], st[1][r]), fmaxf(st[2][r], st[3][r]));
      mx = hmax16(mx);
      float mnew  = fmaxf(mst[r], mx);
      float alpha = __expf(mst[r] - mnew);
      mst[r] = mnew;
      float ls = 0.0f;
#pragma unroll
      for (int nt = 0; nt < 4; ++nt) {
        float p = __expf(st[nt][r] - mnew);
        pr[nt][r] = p;
        ls += p;
      }
      ls = hsum16(ls);
      lst[r] = lst[r] * alpha + ls;
#pragma unroll
      for (int dt = 0; dt < 4; ++dt) oacc[dt][r] *= alpha;
    }

    // Relayout P (C-layout -> A-layout) via wave-private LDS scratch.
    __bf16* pb = ldsP[wave];
#pragma unroll
    for (int nt = 0; nt < 4; ++nt)
#pragma unroll
      for (int r = 0; r < 8; ++r)
        pb[(r + 8 * hl) * 72 + nt * 16 + l16] = (__bf16)pr[nt][r];
    asm volatile("s_wait_dscnt 0" ::: "memory");  // wave's DS ops are in-order
    v16bf ap[2];
#pragma unroll
    for (int j = 0; j < 2; ++j)
      ap[j] = frag_ld(pb + l16 * 72 + 32 * j + 8 * hl);

    // O += P @ V (B column n = dh index -> Vt row n, kv-contiguous).
#pragma unroll
    for (int dt = 0; dt < 4; ++dt)
#pragma unroll
      for (int j = 0; j < 2; ++j) {
        v16bf bv = frag_ld(&Vt[(dt * 16 + l16) * 72 + 32 * j + 8 * hl]);
        oacc[dt] = wmma_bf16(ap[j], bv, oacc[dt]);
      }
  }

  // Epilogue: O /= l -> (4096, 1024) fp32.
#pragma unroll
  for (int r = 0; r < 8; ++r) {
    float inv = 1.0f / lst[r];
    size_t row = srow0 + q0 + 8 * hl + r;
#pragma unroll
    for (int dt = 0; dt < 4; ++dt)
      out[row * 1024 + head * 64 + dt * 16 + l16] = oacc[dt][r] * inv;
  }
}

// ---------------------------------------------------------------------------
extern "C" void kernel_launch(void* const* d_in, const int* in_sizes, int n_in,
                              void* d_out, int out_size, void* d_ws,
                              size_t ws_size, hipStream_t stream) {
  (void)in_sizes; (void)n_in; (void)out_size; (void)ws_size;
  const float* x     = (const float*)d_in[0];
  const float* w_in  = (const float*)d_in[1];
  const float* b_in  = (const float*)d_in[2];
  const float* w_out = (const float*)d_in[3];
  const float* b_out = (const float*)d_in[4];
  float* out = (float*)d_out;

  __bf16* qk  = (__bf16*)d_ws;                      // 4096 x 2048 bf16 (16MB)
  __bf16* vt  = qk + (size_t)4096 * 2048;           // 32 x [64][2048] bf16 (8MB)
  float* attn = (float*)(vt + (size_t)4096 * 1024); // 4096 x 1024 fp32 (16MB)

  dim3 blk(256);
  gemm_qkv_bf16<<<dim3(3072 / 128, 4096 / 128), blk, 0, stream>>>(
      x, w_in, b_in, qk, vt);
  flash_attn<<<dim3(2048 / 128, 32), blk, 0, stream>>>(qk, vt, attn);
  gemm_out_f32<<<dim3(1024 / 128, 4096 / 128), blk, 0, stream>>>(
      attn, w_out, b_out, out);
}
